// ModEdgeConv_11630771437590
// MI455X (gfx1250) — matrix operand order repaired and checked by reading, not compiled, not run on hardware
//
#include <hip/hip_runtime.h>
#include <hip/hip_bf16.h>

typedef __attribute__((ext_vector_type(16))) _Float16 v16h;
typedef __attribute__((ext_vector_type(8)))  float    v8f;

#define B_ 4
#define F_ 3
#define N_ 8192
#define D_ 16
#define K_ 32
#define O_ 64
#define ALPHA_ 0.2f
#define TILE_ 256

// ---------------------------------------------------------------------------
// Kernel 1: transpose x (B,D,N) -> xt (B,N,D) so channel vectors are contiguous
// ---------------------------------------------------------------------------
__global__ __launch_bounds__(256)
void xt_kernel(const float* __restrict__ x, float* __restrict__ xt) {
  int t = blockIdx.x * 256 + threadIdx.x;     // t enumerates (b, n, c)
  int c = t & (D_ - 1);
  int n = (t >> 4) & (N_ - 1);
  int b = t >> 17;                            // 4 + 13 bits
  xt[t] = x[((size_t)b * D_ + c) * N_ + n];
}

// ---------------------------------------------------------------------------
// Kernel 2: fused KNN (no 1GB distance matrix to HBM).
// 256 queries per block (all same batch since 256 | 8192). Candidate points
// staged through LDS in tiles; per-thread top-K list lives in LDS with the
// current worst cached in registers for O(1) rejection.
// ---------------------------------------------------------------------------
__global__ __launch_bounds__(256)
void knn_kernel(const float* __restrict__ pts, int* __restrict__ idxout) {
  __shared__ float sx[TILE_], sy[TILE_], sz[TILE_];
  __shared__ float sdist[K_ * 256];
  __shared__ int   sind [K_ * 256];

  int tid = threadIdx.x;
  int bn  = blockIdx.x * 256 + tid;
  int b   = bn >> 13;                         // / N_
  int n   = bn & (N_ - 1);
  const float* pb = pts + (size_t)b * 3 * N_;

  float px = pb[n], py = pb[N_ + n], pz = pb[2 * N_ + n];

  for (int q = 0; q < K_; ++q) sdist[q * 256 + tid] = 3.4e38f;
  float worstd = 3.4e38f;
  int   worsts = 0;

  for (int m0 = 0; m0 < N_; m0 += TILE_) {
    __syncthreads();
    sx[tid] = pb[m0 + tid];
    sy[tid] = pb[N_ + m0 + tid];
    sz[tid] = pb[2 * N_ + m0 + tid];
    __syncthreads();

    for (int i = 0; i < TILE_; ++i) {
      float dx = px - sx[i], dy = py - sy[i], dz = pz - sz[i];
      float d = dx * dx + dy * dy + dz * dz;   // squared distance (== -neg_dist)
      if (d < worstd) {
        sdist[worsts * 256 + tid] = d;
        sind [worsts * 256 + tid] = m0 + i;
        float wd = -1.0f; int wsl = 0;
        for (int q = 0; q < K_; ++q) {
          float dv = sdist[q * 256 + tid];
          if (dv > wd) { wd = dv; wsl = q; }
        }
        worstd = wd; worsts = wsl;
      }
    }
  }

  int* op = idxout + (size_t)bn * K_;
  for (int q = 0; q < K_; ++q) op[q] = sind[q * 256 + tid];
}

// ---------------------------------------------------------------------------
// Kernel 3: gather + edge GEMM via v_wmma_f32_16x16x32_f16 + LeakyReLU + k-mean.
// One wave per point n. A tile (16x32 f16, per ISA layout): rows = 16 k-values,
// K-axis = 32 channels (c<16: feat-center, c>=16: center). B tiles = W slices
// (32x16). D layout: lane holds 8 M(=k) rows for one N(=o) channel; LeakyReLU
// per element, in-lane sum + shfl_xor(16) completes the sum over all 32 k.
// ---------------------------------------------------------------------------
__global__ __launch_bounds__(256)
void edgeconv_kernel(const float* __restrict__ xt, const int* __restrict__ idxb,
                     const float* __restrict__ W, float* __restrict__ out) {
  int lane = threadIdx.x & 31;
  int wave = threadIdx.x >> 5;
  int bn   = blockIdx.x * 8 + wave;
  int b    = bn >> 13;                        // / N_
  int n    = bn & (N_ - 1);
  int h    = lane >> 4;                       // lane half: selects K-halves
  int l    = lane & 15;

  // --- B matrices: 4 o-tiles of W (K=32 x N=16), B layout:
  // lane l holds column o = 16*ot + l; element e of v16h = W[o][8*h + (e&7) + 16*(e>>3)].
  v16h bm[4];
  #pragma unroll
  for (int ot = 0; ot < 4; ++ot) {
    const float* wr = W + (size_t)(ot * 16 + l) * (2 * D_) + 8 * h;
    #pragma unroll
    for (int e = 0; e < 16; ++e) bm[ot][e] = (_Float16)wr[(e & 7) + ((e >> 3) << 4)];
  }

  // --- center channels 8h..8h+7 (two float4 loads, 64B-aligned base)
  const float4* cc = (const float4*)(xt + (size_t)bn * D_ + 8 * h);
  float4 c0 = cc[0], c1 = cc[1];

  float sums[4] = {0.f, 0.f, 0.f, 0.f};

  #pragma unroll
  for (int t = 0; t < 2; ++t) {               // two k-tiles: k = 16t + l
    int kr = t * 16 + l;
    int gi = idxb[(size_t)bn * K_ + kr];
    const float4* ff = (const float4*)(xt + ((size_t)b * N_ + gi) * D_ + 8 * h);
    float4 f0 = ff[0], f1 = ff[1];

    // A layout (16-bit 16x32): VGPR j<4 -> c = 8h+2j{,+1} (feat-center),
    //                          VGPR j>=4 -> c = 16 + 8h + 2(j-4){,+1} (center).
    v16h a;
    a[0]  = (_Float16)(f0.x - c0.x); a[1]  = (_Float16)(f0.y - c0.y);
    a[2]  = (_Float16)(f0.z - c0.z); a[3]  = (_Float16)(f0.w - c0.w);
    a[4]  = (_Float16)(f1.x - c1.x); a[5]  = (_Float16)(f1.y - c1.y);
    a[6]  = (_Float16)(f1.z - c1.z); a[7]  = (_Float16)(f1.w - c1.w);
    a[8]  = (_Float16)c0.x; a[9]  = (_Float16)c0.y;
    a[10] = (_Float16)c0.z; a[11] = (_Float16)c0.w;
    a[12] = (_Float16)c1.x; a[13] = (_Float16)c1.y;
    a[14] = (_Float16)c1.z; a[15] = (_Float16)c1.w;

    #pragma unroll
    for (int ot = 0; ot < 4; ++ot) {
      v8f c = {};
      // C must be 0: LeakyReLU happens per (k,o) BEFORE the k-reduction.
      v8f d = __builtin_amdgcn_wmma_f32_16x16x32_f16(
          /*neg_a=*/false, a, /*neg_b=*/false, bm[ot],
          /*c_mod=*/(short)0, c, /*reuse_a=*/false, /*reuse_b=*/false);
      float s = 0.f;
      #pragma unroll
      for (int r = 0; r < 8; ++r) {
        float v = d[r];
        s += (v >= 0.f) ? v : ALPHA_ * v;     // LeakyReLU, then sum 8 k-rows
      }
      sums[ot] += s;
    }
  }

  // Combine lane halves: lanes 0-15 hold k rows {16t+0..7}, lanes 16-31 {16t+8..15}
  #pragma unroll
  for (int ot = 0; ot < 4; ++ot) sums[ot] += __shfl_xor(sums[ot], 16, 32);

  if (lane < 16) {
    const float scl = 1.0f / (float)K_;
    size_t base = (size_t)b * O_ * N_ + n;
    #pragma unroll
    for (int ot = 0; ot < 4; ++ot)
      out[base + (size_t)(ot * 16 + l) * N_] = sums[ot] * scl;
  }
}

// ---------------------------------------------------------------------------
extern "C" void kernel_launch(void* const* d_in, const int* in_sizes, int n_in,
                              void* d_out, int out_size, void* d_ws, size_t ws_size,
                              hipStream_t stream) {
  const float* points = (const float*)d_in[0];   // (B, F, N)
  const float* x      = (const float*)d_in[1];   // (B, D, N)
  const float* W      = (const float*)d_in[2];   // (O, 2D)
  float* out = (float*)d_out;                    // (B, O, N)

  float* xt   = (float*)d_ws;                                          // 2 MB
  int*   idxb = (int*)((char*)d_ws + (size_t)B_ * N_ * D_ * sizeof(float)); // 4 MB

  xt_kernel      <<<(B_ * N_ * D_) / 256, 256, 0, stream>>>(x, xt);
  knn_kernel     <<<(B_ * N_) / 256,      256, 0, stream>>>(points, idxb);
  edgeconv_kernel<<<(B_ * N_) / 8,        256, 0, stream>>>(xt, idxb, W, out);
}
